// AdaptiveFeatureLinkedCosineLoss_40303973105851
// MI455X (gfx1250) — matrix operand order repaired and checked by reading, not compiled, not run on hardware
//
#include <hip/hip_runtime.h>
#include <hip/hip_bf16.h>

// ---------------------------------------------------------------------------
// AdaptiveFeatureLinkedCosineLoss for MI455X (gfx1250, wave32, WMMA)
//
// cos_sim[b] = sum_ij rn[b,i] * link_n[i,j] * an[b,j]
//            = rowwise_dot( rn @ link_n , an )          -> bf16 WMMA GEMM
// loss = -mean(cos_sim) / tau(temp_param, mean entropies)
//
// WMMA kernel uses 2x2 register tiling: 2 A-fragments x 2 B-fragments per
// K-chunk -> 4 v_wmma per 8 b128 loads (2 loads/WMMA instead of 4).
// ---------------------------------------------------------------------------

typedef __attribute__((ext_vector_type(16))) __bf16 v16bf;
typedef __attribute__((ext_vector_type(8)))  __bf16 v8bf;
typedef __attribute__((ext_vector_type(8)))  float  v8f;

#define B_ROWS 8192
#define DDIM   1024

__device__ __forceinline__ unsigned short f32_to_bf16_rne(float f) {
    unsigned int u = __float_as_uint(f);
    u += 0x7FFFu + ((u >> 16) & 1u);          // round-to-nearest-even
    return (unsigned short)(u >> 16);
}
__device__ __forceinline__ float bf16_bits_to_f32(unsigned short h) {
    return __uint_as_float(((unsigned int)h) << 16);
}

// A 16x32 bf16 fragment: lane<16 -> row=lane, K {0..15};
// lane>=16 -> row=lane-16, K {16..31}.  ka0/ka1 are the per-half K bases.
__device__ __forceinline__ v16bf load_frag_a(const __bf16* base, size_t off,
                                             int ka0, int ka1) {
    v8bf lo = *reinterpret_cast<const v8bf*>(base + off + ka0);
    v8bf hi = *reinterpret_cast<const v8bf*>(base + off + ka1);
    return __builtin_shufflevector(lo, hi,
                                   0,1,2,3,4,5,6,7,8,9,10,11,12,13,14,15);
}
// B 32x16 bf16 fragment: lane l -> col l%16, K {0..15} (l<16) / {16..31},
// contiguous 2 K per VGPR -> one contiguous 32B run of a linkT row.
__device__ __forceinline__ v16bf load_frag_b(const __bf16* base, size_t off) {
    v8bf lo = *reinterpret_cast<const v8bf*>(base + off);
    v8bf hi = *reinterpret_cast<const v8bf*>(base + off + 8);
    return __builtin_shufflevector(lo, hi,
                                   0,1,2,3,4,5,6,7,8,9,10,11,12,13,14,15);
}

// ---------------------------------------------------------------------------
// Zero the 4 accumulator floats at the head of the workspace.
// ---------------------------------------------------------------------------
__global__ void zero_accum_kernel(float* a) {
    if (threadIdx.x < 4) a[threadIdx.x] = 0.0f;
}

// ---------------------------------------------------------------------------
// Row L2-normalize a (rows x 1024) f32 matrix -> bf16, and accumulate the
// total row entropy sum (-sum v*log(v+1e-8)) into *ent_accum.
// One 256-thread block per row, 4 elements per thread.
// ---------------------------------------------------------------------------
__global__ void __launch_bounds__(256)
norm_rows_entropy_kernel(const float* __restrict__ x,
                         unsigned short* __restrict__ out_bf16,
                         float* __restrict__ ent_accum) {
    __shared__ float red[8];
    const int row = blockIdx.x;
    const int tid = threadIdx.x;
    const float* xr = x + (size_t)row * DDIM;

    float v[4];
    float ss = 0.0f;
#pragma unroll
    for (int i = 0; i < 4; ++i) {
        v[i] = xr[tid + 256 * i];
        ss += v[i] * v[i];
    }
    for (int m = 16; m; m >>= 1) ss += __shfl_xor(ss, m, 32);
    if ((tid & 31) == 0) red[tid >> 5] = ss;
    __syncthreads();
    float tot = 0.0f;
#pragma unroll
    for (int i = 0; i < 8; ++i) tot += red[i];

    const float inv = 1.0f / fmaxf(sqrtf(tot), 1e-12f);

    float ent = 0.0f;
#pragma unroll
    for (int i = 0; i < 4; ++i) {
        float nv = v[i] * inv;
        out_bf16[(size_t)row * DDIM + tid + 256 * i] = f32_to_bf16_rne(nv);
        ent += -nv * __logf(nv + 1e-8f);
    }
    __syncthreads();                 // everyone done reading red[]
    for (int m = 16; m; m >>= 1) ent += __shfl_xor(ent, m, 32);
    if ((tid & 31) == 0) red[tid >> 5] = ent;
    __syncthreads();
    if (tid == 0) {
        float e = 0.0f;
#pragma unroll
        for (int i = 0; i < 8; ++i) e += red[i];
        atomicAdd(ent_accum, e);
    }
}

// ---------------------------------------------------------------------------
// Row L2-normalize link (1024x1024) and store TRANSPOSED bf16:
// linkT[j*D + i] = link_n[i][j].  One block per source row i.
// ---------------------------------------------------------------------------
__global__ void __launch_bounds__(256)
norm_link_transpose_kernel(const float* __restrict__ link,
                           unsigned short* __restrict__ linkT) {
    __shared__ float red[8];
    const int row = blockIdx.x;
    const int tid = threadIdx.x;
    const float* xr = link + (size_t)row * DDIM;

    float v[4];
    float ss = 0.0f;
#pragma unroll
    for (int i = 0; i < 4; ++i) {
        v[i] = xr[tid + 256 * i];
        ss += v[i] * v[i];
    }
    for (int m = 16; m; m >>= 1) ss += __shfl_xor(ss, m, 32);
    if ((tid & 31) == 0) red[tid >> 5] = ss;
    __syncthreads();
    float tot = 0.0f;
#pragma unroll
    for (int i = 0; i < 8; ++i) tot += red[i];

    const float inv = 1.0f / fmaxf(sqrtf(tot), 1e-12f);
#pragma unroll
    for (int i = 0; i < 4; ++i) {
        const int col = tid + 256 * i;
        linkT[(size_t)col * DDIM + row] = f32_to_bf16_rne(v[i] * inv);
    }
}

// ---------------------------------------------------------------------------
// Bilinear contraction on WMMA, 2x2 register-tiled.
// Each wave owns 32 rows (2 row-blocks of 16).  Outer loop walks 32-column
// tiles of link (2 n-tiles of 16).  Inner loop: per 32-K chunk load 2 A frags
// + 2 B frags (8 b128) and issue 4 v_wmma_f32_16x16x32_bf16 -> acc00..acc11.
// After each column tile, fold acc * an into a per-lane scalar; wave-reduce
// and atomicAdd once at the end.
//
//   C 16x16 f32: lane l -> col l%16, VGPR v -> row v + 8*(l>=16)
// ---------------------------------------------------------------------------
__global__ void __launch_bounds__(256)
bilinear_wmma_kernel(const unsigned short* __restrict__ rnb,
                     const unsigned short* __restrict__ anb,
                     const unsigned short* __restrict__ linkTb,
                     float* __restrict__ cos_accum) {
    const int lane = threadIdx.x & 31;
    const int wave = threadIdx.x >> 5;
    const int grp  = blockIdx.x * 8 + wave;    // 0..255 (8192/32 row groups)
    const int rbase = grp * 32;
    const int half = lane >> 4;                // 0: lanes 0-15, 1: lanes 16-31
    const int lr   = lane & 15;

    const __bf16* rn = reinterpret_cast<const __bf16*>(rnb);
    const __bf16* lt = reinterpret_cast<const __bf16*>(linkTb);

    const size_t arow0 = (size_t)(rbase + lr) * DDIM;        // A rows, block 0
    const size_t arow1 = (size_t)(rbase + 16 + lr) * DDIM;   // A rows, block 1
    const int ka0 = half ? 16 : 0;
    const int ka1 = half ? 24 : 8;

    float local = 0.0f;

    for (int nt = 0; nt < DDIM / 32; ++nt) {
        const size_t b0off = (size_t)(nt * 32 + lr) * DDIM + half * 16;
        const size_t b1off = (size_t)(nt * 32 + 16 + lr) * DDIM + half * 16;
        v8f acc00 = {}, acc01 = {}, acc10 = {}, acc11 = {};
#pragma unroll 2
        for (int kt = 0; kt < DDIM; kt += 32) {
            v16bf a0 = load_frag_a(rn, arow0 + kt, ka0, ka1);
            v16bf a1 = load_frag_a(rn, arow1 + kt, ka0, ka1);
            v16bf b0 = load_frag_b(lt, b0off + kt);
            v16bf b1 = load_frag_b(lt, b1off + kt);
            acc00 = __builtin_amdgcn_wmma_f32_16x16x32_bf16(
                        false, a0, false, b0, (short)0, acc00, false, false);
            acc01 = __builtin_amdgcn_wmma_f32_16x16x32_bf16(
                        false, a0, false, b1, (short)0, acc01, false, false);
            acc10 = __builtin_amdgcn_wmma_f32_16x16x32_bf16(
                        false, a1, false, b0, (short)0, acc10, false, false);
            acc11 = __builtin_amdgcn_wmma_f32_16x16x32_bf16(
                        false, a1, false, b1, (short)0, acc11, false, false);
        }
        // epilogue: C[m,n] * an[row, col]
#pragma unroll
        for (int v = 0; v < 8; ++v) {
            const int m = v + 8 * half;
            const int n0 = nt * 32 + lr;
            const float an00 = bf16_bits_to_f32(anb[(size_t)(rbase + m)      * DDIM + n0]);
            const float an01 = bf16_bits_to_f32(anb[(size_t)(rbase + m)      * DDIM + n0 + 16]);
            const float an10 = bf16_bits_to_f32(anb[(size_t)(rbase + 16 + m) * DDIM + n0]);
            const float an11 = bf16_bits_to_f32(anb[(size_t)(rbase + 16 + m) * DDIM + n0 + 16]);
            local += acc00[v] * an00 + acc01[v] * an01
                   + acc10[v] * an10 + acc11[v] * an11;
        }
    }

    for (int m = 16; m; m >>= 1) local += __shfl_xor(local, m, 32);
    if (lane == 0) atomicAdd(cos_accum, local);
}

// ---------------------------------------------------------------------------
// Scalar epilogue: temperature + loss.
// ---------------------------------------------------------------------------
__global__ void finalize_kernel(const float* __restrict__ accum,
                                const float* __restrict__ temp_param,
                                float* __restrict__ out) {
    const float cos_sum  = accum[0];
    const float ent_rna  = accum[1] / (float)B_ROWS;
    const float ent_atac = accum[2] / (float)B_ROWS;
    const float avg_entropy = 0.5f * (ent_rna + ent_atac);
    const float t = temp_param[0];
    const float s = 1.0f / (1.0f + __expf(-t));
    const float scale = s * 0.1f + (1.0f - s) * avg_entropy;
    const float tau = fminf(fmaxf(scale, 0.01f), 1.0f);
    out[0] = -(cos_sum / (float)B_ROWS) / tau;
}

// ---------------------------------------------------------------------------
// Inputs (setup_inputs order): z_rna (B*D f32), z_atac (B*D f32),
// link_matrix (D*D f32), temp_param (1 f32).  Output: 1 f32.
// Workspace layout:
//   [0,16)            : accum{cos_sum, ent_rna_sum, ent_atac_sum, pad}
//   [256, +16MB)      : rn bf16  (B*D)
//   [+16MB, +32MB)    : an bf16  (B*D)
//   [+32MB, +34MB)    : linkT bf16 (D*D)
// ---------------------------------------------------------------------------
extern "C" void kernel_launch(void* const* d_in, const int* in_sizes, int n_in,
                              void* d_out, int out_size, void* d_ws, size_t ws_size,
                              hipStream_t stream) {
    (void)in_sizes; (void)n_in; (void)out_size; (void)ws_size;

    const float* z_rna  = (const float*)d_in[0];
    const float* z_atac = (const float*)d_in[1];
    const float* link   = (const float*)d_in[2];
    const float* temp   = (const float*)d_in[3];

    char* ws = (char*)d_ws;
    float* accum         = (float*)ws;
    unsigned short* rnb  = (unsigned short*)(ws + 256);
    unsigned short* anb  = rnb + (size_t)B_ROWS * DDIM;
    unsigned short* ltb  = anb + (size_t)B_ROWS * DDIM;

    zero_accum_kernel<<<1, 32, 0, stream>>>(accum);
    norm_rows_entropy_kernel<<<B_ROWS, 256, 0, stream>>>(z_rna,  rnb, accum + 1);
    norm_rows_entropy_kernel<<<B_ROWS, 256, 0, stream>>>(z_atac, anb, accum + 2);
    norm_link_transpose_kernel<<<DDIM, 256, 0, stream>>>(link, ltb);
    // 256 row groups of 32 rows; 8 waves/block -> 32 blocks
    bilinear_wmma_kernel<<<32, 256, 0, stream>>>(rnb, anb, ltb, accum);
    finalize_kernel<<<1, 1, 0, stream>>>(accum, temp, (float*)d_out);
}
